// Head_8864812499258
// MI455X (gfx1250) — matrix operand order repaired and checked by reading, not compiled, not run on hardware
//
#include <hip/hip_runtime.h>
#include <hip/hip_bf16.h>

#define Bc 4
#define Tc 4096
#define Ec 32
#define Hc 32

typedef __attribute__((ext_vector_type(16))) __bf16 v16bf;
typedef __attribute__((ext_vector_type(8)))  float  v8f;

union V16 { v16bf v; uint4 q[2]; };

#define WMMA_BF16(A, B, C) \
    __builtin_amdgcn_wmma_f32_16x16x32_bf16(false, (A), false, (B), (short)0, (C), false, false)

// Build an A-operand (16x32, 16-bit) lane fragment from f32 memory.
// Contiguous layout: vector elems 0..15 = K kb..kb+15,
// where `row` already points at &X[rowIdx][kb].
__device__ __forceinline__ v16bf load_cvt_A(const float* __restrict__ row) {
    v16bf v;
#pragma unroll
    for (int i = 0; i < 8; i++) v[i] = (__bf16)row[i];
#pragma unroll
    for (int i = 0; i < 8; i++) v[8 + i] = (__bf16)row[8 + i];
    return v;
}
// B-operand fragment: 16 contiguous K values per lane.
__device__ __forceinline__ v16bf load_cvt_B(const float* __restrict__ row) {
    v16bf v;
#pragma unroll
    for (int i = 0; i < 16; i++) v[i] = (__bf16)row[i];
    return v;
}

// ---------------------------------------------------------------------------
// Kernel 1: q = x·Wq^T, k = x·Wk^T (row-major bf16 [B*T,32]),
//           vT = (x·Wv^T)^T (bf16 [B,32,T]) so PV's B-operand is a row load.
// One wave per 16-row tile of x. 6 WMMAs per wave.
// ---------------------------------------------------------------------------
__global__ __launch_bounds__(256)
void proj_qkv(const float* __restrict__ x,  const float* __restrict__ Wq,
              const float* __restrict__ Wk, const float* __restrict__ Wv,
              __bf16* __restrict__ qo, __bf16* __restrict__ ko,
              __bf16* __restrict__ vto)
{
    const int lane  = threadIdx.x & 31;
    const int wid   = blockIdx.x * (blockDim.x >> 5) + (threadIdx.x >> 5);
    const int b     = wid / (Tc / 16);
    const int tbase = (wid % (Tc / 16)) * 16;
    const int l15   = lane & 15, hig = lane >> 4;
    const float* xb = x + (size_t)b * Tc * Ec;

    // x as A-operand (M=t, K=n) and as B-operand (K=n, N=t)
    V16 XA, XB;
    XA.v = load_cvt_A(xb + (size_t)(tbase + l15) * Ec + hig * 16);
    XB.v = load_cvt_B(xb + (size_t)(tbase + l15) * Ec + hig * 16);

    const v8f zc = {};
#pragma unroll
    for (int ht = 0; ht < 2; ht++) {
        // Wq/Wk as B-operands (B[n][h] = W[h][n] -> load W rows)
        V16 WQB, WKB, WVA;
        WQB.v = load_cvt_B(Wq + (size_t)(ht * 16 + l15) * Ec + hig * 16);
        WKB.v = load_cvt_B(Wk + (size_t)(ht * 16 + l15) * Ec + hig * 16);
        v8f dq = WMMA_BF16(XA.v, WQB.v, zc);
        v8f dk = WMMA_BF16(XA.v, WKB.v, zc);
#pragma unroll
        for (int r = 0; r < 8; r++) {            // D layout: M = r + 8*hig, N = l15
            int t = tbase + r + 8 * hig;
            qo[((size_t)b * Tc + t) * Hc + ht * 16 + l15] = (__bf16)dq[r];
            ko[((size_t)b * Tc + t) * Hc + ht * 16 + l15] = (__bf16)dk[r];
        }

        // vT[h][t] = sum_n Wv[h][n] x[t][n]: A = Wv rows (M=h), B = XB
        WVA.v = load_cvt_A(Wv + (size_t)(ht * 16 + l15) * Ec + hig * 16);
        v8f dv = WMMA_BF16(WVA.v, XB.v, zc);
#pragma unroll
        for (int r = 0; r < 8; r++) {
            int h = ht * 16 + r + 8 * hig;
            vto[((size_t)b * Hc + h) * Tc + tbase + l15] = (__bf16)dv[r];
        }
    }
}

// ---------------------------------------------------------------------------
// Kernel 2: flash attention, one wave per 16-query tile, 32-key tiles,
// causal loop bound, online softmax (log2 domain -> raw v_exp_f32) via
// half-wave shfl_xor reductions, P bounced through LDS with s_wait_dscnt.
// ---------------------------------------------------------------------------
__global__ __launch_bounds__(256)
void flash_attn(const __bf16* __restrict__ qb,
                const __bf16* __restrict__ kb,
                const __bf16* __restrict__ vtb,
                float* __restrict__ out)
{
    __shared__ __bf16 plds[8][16 * 32];           // per-wave 16x32 bf16 P tile
    const int lane  = threadIdx.x & 31;
    const int w     = threadIdx.x >> 5;
    const int wid   = blockIdx.x * 8 + w;
    const int b     = wid / (Tc / 16);
    const int qbase = (wid % (Tc / 16)) * 16;
    const int l15   = lane & 15, hig = lane >> 4;
    const float scale2 = 0.2550348597f;           // 32^-0.5 * log2(e)

    // Q tile as A-operand (bf16 already: two 16B vector loads)
    V16 QA;
    {
        const __bf16* qr = qb + ((size_t)b * Tc + qbase + l15) * Hc + hig * 16;
        QA.q[0] = *(const uint4*)qr;
        QA.q[1] = *(const uint4*)(qr + 8);
    }

    float m[8], lsum[8];
    int   rowg[8];
    v8f acc0 = {}, acc1 = {};
#pragma unroll
    for (int r = 0; r < 8; r++) {
        rowg[r] = qbase + r + 8 * hig;
        m[r]    = -__builtin_inff();
        lsum[r] = 0.0f;
    }

    const int nkt = (qbase + 16 + 31) >> 5;       // causal: keys 0..qbase+15
    for (int kt = 0; kt < nkt; ++kt) {
        const int k0 = kt * 32;

        // K tiles as B-operands: lane holds key row (l15), head chunk hig*16..+15
        V16 KB0, KB1;
        {
            const __bf16* kr = kb + ((size_t)b * Tc + k0 + l15) * Hc + hig * 16;
            KB0.q[0] = *(const uint4*)kr;
            KB0.q[1] = *(const uint4*)(kr + 8);
            const __bf16* kr1 = kr + 16 * Hc;
            KB1.q[0] = *(const uint4*)kr1;
            KB1.q[1] = *(const uint4*)(kr1 + 8);
        }
        const v8f zc = {};
        v8f s0 = WMMA_BF16(QA.v, KB0.v, zc);      // scores cols k0..k0+15
        v8f s1 = WMMA_BF16(QA.v, KB1.v, zc);      // scores cols k0+16..k0+31

#pragma unroll
        for (int r = 0; r < 8; r++) {
            // log2-domain scaled scores with causal mask
            float a = (k0 + l15      <= rowg[r]) ? s0[r] * scale2 : -__builtin_inff();
            float c = (k0 + 16 + l15 <= rowg[r]) ? s1[r] * scale2 : -__builtin_inff();
            // row max across the 16 lanes holding this row
            float t = fmaxf(a, c);
            t = fmaxf(t, __shfl_xor(t, 1));
            t = fmaxf(t, __shfl_xor(t, 2));
            t = fmaxf(t, __shfl_xor(t, 4));
            t = fmaxf(t, __shfl_xor(t, 8));
            float mn = fmaxf(m[r], t);
            float pr = exp2f(m[r] - mn);          // v_exp_f32 directly
            m[r] = mn;
            float p0 = exp2f(a - mn);
            float p1 = exp2f(c - mn);
            __bf16* pp = &plds[w][(r + 8 * hig) * 32];
            pp[l15]      = (__bf16)p0;
            pp[16 + l15] = (__bf16)p1;
            float ts = p0 + p1;
            ts += __shfl_xor(ts, 1);
            ts += __shfl_xor(ts, 2);
            ts += __shfl_xor(ts, 4);
            ts += __shfl_xor(ts, 8);
            lsum[r] = lsum[r] * pr + ts;
            acc0[r] *= pr;
            acc1[r] *= pr;
        }
        asm volatile("s_wait_dscnt 0" ::: "memory");

        // reload P in A-operand layout from LDS
        V16 PA;
        {
            const __bf16* pr_ = &plds[w][l15 * 32] + hig * 16;
            PA.q[0] = *(const uint4*)pr_;
            PA.q[1] = *(const uint4*)(pr_ + 8);
        }
        // V tiles as B-operands from vT: lane holds head col l15, key run k0+hig*16..+15
        V16 VB0, VB1;
        {
            const __bf16* vr = vtb + ((size_t)b * Hc + l15) * Tc + k0 + hig * 16;
            VB0.q[0] = *(const uint4*)vr;
            VB0.q[1] = *(const uint4*)(vr + 8);
            const __bf16* vr1 = vr + 16 * Tc;
            VB1.q[0] = *(const uint4*)vr1;
            VB1.q[1] = *(const uint4*)(vr1 + 8);
        }
        acc0 = WMMA_BF16(PA.v, VB0.v, acc0);      // out cols 0..15
        acc1 = WMMA_BF16(PA.v, VB1.v, acc1);      // out cols 16..31
    }

#pragma unroll
    for (int r = 0; r < 8; r++) {
        float inv = 1.0f / lsum[r];
        float* orow = out + ((size_t)b * Tc + rowg[r]) * Hc;
        orow[l15]      = acc0[r] * inv;
        orow[16 + l15] = acc1[r] * inv;
    }
}

extern "C" void kernel_launch(void* const* d_in, const int* in_sizes, int n_in,
                              void* d_out, int out_size, void* d_ws, size_t ws_size,
                              hipStream_t stream)
{
    const float* x  = (const float*)d_in[0];
    const float* Wq = (const float*)d_in[1];
    const float* Wk = (const float*)d_in[2];
    const float* Wv = (const float*)d_in[3];

    const size_t bytesPer = (size_t)Bc * Tc * Hc * sizeof(__bf16); // 1 MiB each
    __bf16* qo  = (__bf16*)d_ws;
    __bf16* ko  = (__bf16*)((char*)d_ws + bytesPer);
    __bf16* vto = (__bf16*)((char*)d_ws + 2 * bytesPer);
    float* out = (float*)d_out;

    dim3 blk(256);                                // 8 waves (wave32)
    dim3 grd((Bc * Tc / 16) / 8);                 // 1024 waves total
    proj_qkv<<<grd, blk, 0, stream>>>(x, Wq, Wk, Wv, qo, ko, vto);
    flash_attn<<<grd, blk, 0, stream>>>(qo, ko, vto, out);
}
